// MultiHeadAttention_50534585205084
// MI455X (gfx1250) — matrix operand order repaired and checked by reading, not compiled, not run on hardware
//
#include <hip/hip_runtime.h>
#include <hip/hip_bf16.h>

// Problem constants (from reference)
#define Bsz   4
#define Sn    2048
#define Dm    1024
#define Hn    16
#define KPn   8
#define DDn   64
#define Npool 256
#define Mrows (Bsz*Sn)             // 8192
#define NORM  0.3535533905932738f  // 64^-0.25

typedef __attribute__((ext_vector_type(16))) __bf16 v16bf;
typedef __attribute__((ext_vector_type(8)))  __bf16 v8bf;
typedef __attribute__((ext_vector_type(4)))  __bf16 v4bf;
typedef __attribute__((ext_vector_type(8)))  float  v8f;
typedef __attribute__((ext_vector_type(4)))  int    v4i;

union V16U { v16bf v; v8bf h[2]; };

// ---------------------------------------------------------------------------
// CDNA5 async global->LDS copy (ASYNCcnt path), guarded so compile never breaks
// Builtin signature (from clang diagnostic): (as1 v4i* src, as3 v4i* dst, Ii, Ii)
// ---------------------------------------------------------------------------
#if defined(__gfx1250__) && __has_builtin(__builtin_amdgcn_global_load_async_to_lds_b128)
#define USE_ASYNC_LDS 1
#else
#define USE_ASYNC_LDS 0
#endif

__device__ __forceinline__ void g2l_b128(__bf16* lds, const __bf16* g) {
#if USE_ASYNC_LDS
  __builtin_amdgcn_global_load_async_to_lds_b128(
      (__attribute__((address_space(1))) v4i*)(g),
      (__attribute__((address_space(3))) v4i*)(lds), 0, 0);
#else
  *(v8bf*)lds = *(const v8bf*)g;   // fallback: VGPR-staged 16B copy
#endif
}

__device__ __forceinline__ void wait_async_copies() {
#if USE_ASYNC_LDS
#if __has_builtin(__builtin_amdgcn_s_wait_asynccnt)
  __builtin_amdgcn_s_wait_asynccnt(0);
#else
  asm volatile("s_wait_asynccnt 0" ::: "memory");
#endif
#endif
}

// ---------------------------------------------------------------------------
// GEMM: out[M=8192,N=1024] = (A[M,K]bf16 @ Wt[N,K]bf16^T + bias) * scale
// A row-major [m][k], Wt pre-transposed row-major [n][k]: both tiles stage as
// contiguous B128 chunks (async, double-buffered) in the exact WMMA fragment
// layout. 256 thr = 8 waves (2x4); macrotile 128x128, BK=32; 8 wmma/K-step.
// ---------------------------------------------------------------------------
__global__ __launch_bounds__(256) void gemm_bf16_wmma(
    const __bf16* __restrict__ A, const __bf16* __restrict__ Wt,
    const float* __restrict__ bias, float scale, float* __restrict__ out)
{
  __shared__ alignas(32) __bf16 As[2][128 * 32];   // [m][k]
  __shared__ alignas(32) __bf16 Bs[2][128 * 32];   // [n][k]

  const int tid   = threadIdx.x;
  const int lane  = tid & 31;
  const int wave  = tid >> 5;
  const int waveM = wave >> 2;   // 0..1
  const int waveN = wave & 3;    // 0..3
  const int half  = lane >> 4;   // 0..1
  const int l16   = lane & 15;

  const int nBase = blockIdx.x * 128;
  const int mBase = blockIdx.y * 128;

  // Staging: 512 chunks of 16B per tile; each thread copies chunks tid, tid+256
  const int c0 = tid,        c1 = tid + 256;
  const int r0 = c0 >> 2,    k0 = (c0 & 3) * 8;
  const int r1 = c1 >> 2,    k1 = (c1 & 3) * 8;

  v8f acc[4][2];
#pragma unroll
  for (int i = 0; i < 4; ++i)
#pragma unroll
    for (int j = 0; j < 2; ++j) acc[i][j] = (v8f){};

  // Prologue: stream tile 0 into buffer 0
  g2l_b128(&As[0][r0 * 32 + k0], &A [(size_t)(mBase + r0) * Dm + k0]);
  g2l_b128(&As[0][r1 * 32 + k1], &A [(size_t)(mBase + r1) * Dm + k1]);
  g2l_b128(&Bs[0][r0 * 32 + k0], &Wt[(size_t)(nBase + r0) * Dm + k0]);
  g2l_b128(&Bs[0][r1 * 32 + k1], &Wt[(size_t)(nBase + r1) * Dm + k1]);

  const int T = Dm / 32;   // 32 K-tiles
  for (int t = 0; t < T; ++t) {
    wait_async_copies();   // this wave's LDS writes landed
    __syncthreads();       // all waves' writes visible; prev reads done

    const int cur = t & 1, nxt = cur ^ 1;
    if (t + 1 < T) {       // stream next tile while computing this one
      const int kb = (t + 1) * 32;
      g2l_b128(&As[nxt][r0 * 32 + k0], &A [(size_t)(mBase + r0) * Dm + kb + k0]);
      g2l_b128(&As[nxt][r1 * 32 + k1], &A [(size_t)(mBase + r1) * Dm + kb + k1]);
      g2l_b128(&Bs[nxt][r0 * 32 + k0], &Wt[(size_t)(nBase + r0) * Dm + kb + k0]);
      g2l_b128(&Bs[nxt][r1 * 32 + k1], &Wt[(size_t)(nBase + r1) * Dm + kb + k1]);
    }

    // B fragments: lane l16 = column; lanes 0-15 K0-15, lanes 16-31 K16-31
    v16bf bfrag[2];
#pragma unroll
    for (int j = 0; j < 2; ++j) {
      int coln = waveN * 32 + j * 16 + l16;
      bfrag[j] = *(const v16bf*)&Bs[cur][coln * 32 + half * 16];
    }
    // A fragments: lane l16 = row; lanes 0-15 K0-15, lanes 16-31 K16-31
#pragma unroll
    for (int i = 0; i < 4; ++i) {
      int row = waveM * 64 + i * 16 + l16;
      V16U a;
      a.h[0] = *(const v8bf*)&As[cur][row * 32 + half * 16];
      a.h[1] = *(const v8bf*)&As[cur][row * 32 + half * 16 + 8];
#pragma unroll
      for (int j = 0; j < 2; ++j) {
        acc[i][j] = __builtin_amdgcn_wmma_f32_16x16x32_bf16(
            false, a.v, false, bfrag[j], (short)0, acc[i][j], false, false);
      }
    }
  }

  // Epilogue: C layout -> VGPR r: M = tile + half*8 + r, N = tile + l16
#pragma unroll
  for (int i = 0; i < 4; ++i) {
    int rowT = mBase + waveM * 64 + i * 16 + half * 8;
#pragma unroll
    for (int j = 0; j < 2; ++j) {
      int col = nBase + waveN * 32 + j * 16 + l16;
      float bv = bias[col];
#pragma unroll
      for (int r = 0; r < 8; ++r) {
        out[(size_t)(rowT + r) * Dm + col] = (acc[i][j][r] + bv) * scale;
      }
    }
  }
}

// ---------------------------------------------------------------------------
// Elementwise fp32 -> bf16 cast (vectorized 4-wide)
// ---------------------------------------------------------------------------
__global__ __launch_bounds__(256) void cast_bf16(
    const float* __restrict__ src, __bf16* __restrict__ dst, int n4)
{
  int i = blockIdx.x * 256 + threadIdx.x;
  if (i >= n4) return;
  float4 v = ((const float4*)src)[i];
  v4bf o = {(__bf16)v.x, (__bf16)v.y, (__bf16)v.z, (__bf16)v.w};
  ((v4bf*)dst)[i] = o;
}

// ---------------------------------------------------------------------------
// Weight transpose + cast: W fp32 [K=1024][N=1024] -> Wt bf16 [N][K]
// 64x64 tiles through padded LDS; coalesced read and write.
// ---------------------------------------------------------------------------
__global__ __launch_bounds__(256) void transpose_cast_bf16(
    const float* __restrict__ W, __bf16* __restrict__ Wt)
{
  __shared__ float tile[64][65];
  const int tid   = threadIdx.x;
  const int nBase = blockIdx.x * 64;
  const int kBase = blockIdx.y * 64;
#pragma unroll
  for (int it = 0; it < 4; ++it) {
    int e = (tid + it * 256) * 4;
    int r = e >> 6;   // k row
    int c = e & 63;   // n col
    float4 v = *(const float4*)&W[(size_t)(kBase + r) * Dm + nBase + c];
    tile[r][c] = v.x; tile[r][c + 1] = v.y; tile[r][c + 2] = v.z; tile[r][c + 3] = v.w;
  }
  __syncthreads();
#pragma unroll
  for (int it = 0; it < 4; ++it) {
    int e  = (tid + it * 256) * 4;
    int rn = e >> 6;  // n row (out)
    int ck = e & 63;  // k col (out)
    v4bf o = {(__bf16)tile[ck][rn], (__bf16)tile[ck + 1][rn],
              (__bf16)tile[ck + 2][rn], (__bf16)tile[ck + 3][rn]};
    *(v4bf*)&Wt[(size_t)(nBase + rn) * Dm + kBase + ck] = o;
  }
}

// ---------------------------------------------------------------------------
// Fused causal depthwise conv (DK=3) + causal avg-pool (window=stride=8).
// x: [B,S,D] fp32 -> outp: [B,H,256,64] fp32 (head-split layout)
// ---------------------------------------------------------------------------
__global__ __launch_bounds__(256) void convpool(
    const float* __restrict__ x, const float* __restrict__ w,
    const float* __restrict__ bias, float* __restrict__ outp)
{
  int idx = blockIdx.x * 256 + threadIdx.x;   // over B*Npool*D
  int d  = idx & (Dm - 1);
  int t2 = idx >> 10;
  int j  = t2 & (Npool - 1);
  int b  = t2 >> 8;

  float w0 = w[d], w1 = w[Dm + d], w2 = w[2 * Dm + d];
  float bc = bias[d];
  const float* xb = x + (size_t)b * Sn * Dm;

  float s = 0.f;
  int t0 = j * KPn - (KPn - 1);
#pragma unroll
  for (int i = 0; i < KPn; ++i) {
    int t = t0 + i;
    if (t < 0) continue;
    float y = bc;
    if (t - 2 >= 0) y += xb[(size_t)(t - 2) * Dm + d] * w0;
    if (t - 1 >= 0) y += xb[(size_t)(t - 1) * Dm + d] * w1;
    y += xb[(size_t)t * Dm + d] * w2;
    s += y;
  }
  int h = d >> 6, dd = d & 63;
  outp[(((size_t)b * Hn + h) * Npool + j) * DDn + dd] = s * (1.0f / KPn);
}

// ---------------------------------------------------------------------------
// Pooled causal attention + fused 64x64 up-projection.
// One block per (b,h); thread tid owns query row tid (n=256).
// K/V reads are lane-uniform (broadcast; 64KB/block, WGP$/L2 resident).
// ---------------------------------------------------------------------------
__global__ __launch_bounds__(256) void attn_pool(
    const float* __restrict__ qp, const float* __restrict__ kp,
    const float* __restrict__ vp, const float* __restrict__ Wup,
    const float* __restrict__ bup, float* __restrict__ upd)
{
  __shared__ float Wls[64 * 64];
  __shared__ float Bls[64];
  const int tid = threadIdx.x;
  const int bh  = blockIdx.x;

#pragma unroll
  for (int it = 0; it < 4; ++it)
    ((float4*)Wls)[tid + it * 256] = ((const float4*)Wup)[tid + it * 256];
  if (tid < 64) Bls[tid] = bup[tid];
  __syncthreads();

  const float* Kb = kp + (size_t)bh * Npool * DDn;
  const float* Vb = vp + (size_t)bh * Npool * DDn;
  const float* qrow = qp + ((size_t)bh * Npool + tid) * DDn;

  float q[DDn];
#pragma unroll
  for (int d = 0; d < DDn; ++d) q[d] = qrow[d];

  const int jq = tid;
  float mx = -1e30f;
  for (int m = 0; m <= jq; ++m) {
    const float* kr = Kb + m * DDn;
    float dot = 0.f;
#pragma unroll
    for (int d = 0; d < DDn; ++d) dot += q[d] * kr[d];
    mx = fmaxf(mx, dot);
  }
  float acc[DDn];
#pragma unroll
  for (int d = 0; d < DDn; ++d) acc[d] = 0.f;
  float sum = 0.f;
  for (int m = 0; m <= jq; ++m) {
    const float* kr = Kb + m * DDn;
    float dot = 0.f;
#pragma unroll
    for (int d = 0; d < DDn; ++d) dot += q[d] * kr[d];
    float p = __expf(dot - mx);
    sum += p;
    const float* vr = Vb + m * DDn;
#pragma unroll
    for (int d = 0; d < DDn; ++d) acc[d] += p * vr[d];
  }
  float inv = 1.f / sum;
#pragma unroll
  for (int d = 0; d < DDn; ++d) acc[d] *= inv;

  float* orow = upd + ((size_t)bh * Npool + jq) * DDn;
#pragma unroll 4
  for (int n = 0; n < DDn; ++n) {
    float o = Bls[n];
#pragma unroll
    for (int d = 0; d < DDn; ++d) o += acc[d] * Wls[d * 64 + n];
    orow[n] = o;
  }
}

// ---------------------------------------------------------------------------
// Repeat-upsample (x8) + head-merge transpose + bf16 cast:
// upd [B,H,256,64] fp32 -> merged [B*S, D] bf16   (A of final GEMM)
// ---------------------------------------------------------------------------
__global__ __launch_bounds__(256) void upsample_merge_bf16(
    const float* __restrict__ upd, __bf16* __restrict__ merged)
{
  int g   = blockIdx.x * 256 + threadIdx.x;  // over (M*D)/4
  int col = (g & 255) * 4;
  int row = g >> 8;
  int b = row >> 11;
  int s = row & 2047;
  int h = col >> 6;
  int dd = col & 63;
  float4 v = *(const float4*)&upd[(((size_t)b * Hn + h) * Npool + (s >> 3)) * DDn + dd];
  v4bf o = {(__bf16)v.x, (__bf16)v.y, (__bf16)v.z, (__bf16)v.w};
  *(v4bf*)&merged[(size_t)row * Dm + col] = o;
}

// ---------------------------------------------------------------------------
extern "C" void kernel_launch(void* const* d_in, const int* in_sizes, int n_in,
                              void* d_out, int out_size, void* d_ws, size_t ws_size,
                              hipStream_t stream) {
  (void)in_sizes; (void)n_in; (void)out_size; (void)ws_size;

  const float* q   = (const float*)d_in[0];
  const float* k   = (const float*)d_in[1];
  const float* v   = (const float*)d_in[2];
  const float* Wq  = (const float*)d_in[3];
  const float* bq  = (const float*)d_in[4];
  const float* Wk  = (const float*)d_in[5];
  const float* bk  = (const float*)d_in[6];
  const float* Wv  = (const float*)d_in[7];
  const float* bv  = (const float*)d_in[8];
  const float* Wup = (const float*)d_in[9];
  const float* bup = (const float*)d_in[10];
  const float* Wc  = (const float*)d_in[11];
  const float* bc  = (const float*)d_in[12];
  const float* wcq = (const float*)d_in[13];
  const float* bcq = (const float*)d_in[14];
  const float* wck = (const float*)d_in[15];
  const float* bck = (const float*)d_in[16];
  const float* wcv = (const float*)d_in[17];
  const float* bcv = (const float*)d_in[18];

  const size_t BSD = (size_t)Bsz * Sn * Dm;           // 8,388,608
  const size_t BHP = (size_t)Bsz * Hn * Npool * DDn;  // 1,048,576
  const size_t DD2 = (size_t)Dm * Dm;                 // 1,048,576

  char* cur = (char*)d_ws;
  float* qh  = (float*)cur; cur += BSD * 4;
  float* kh  = (float*)cur; cur += BSD * 4;
  float* vh  = (float*)cur; cur += BSD * 4;
  float* qp  = (float*)cur; cur += BHP * 4;
  float* kpp = (float*)cur; cur += BHP * 4;
  float* vpp = (float*)cur; cur += BHP * 4;
  float* upd = (float*)cur; cur += BHP * 4;
  __bf16* qb  = (__bf16*)cur; cur += BSD * 2;
  __bf16* kb2 = (__bf16*)cur; cur += BSD * 2;
  __bf16* vb2 = (__bf16*)cur; cur += BSD * 2;
  __bf16* Wqt = (__bf16*)cur; cur += DD2 * 2;
  __bf16* Wkt = (__bf16*)cur; cur += DD2 * 2;
  __bf16* Wvt = (__bf16*)cur; cur += DD2 * 2;
  __bf16* Wct = (__bf16*)cur; cur += DD2 * 2;
  __bf16* mrg = (__bf16*)cur; cur += BSD * 2;

  dim3 blk(256);
  dim3 gemmGrid(Dm / 128, Mrows / 128);   // (8, 64)
  dim3 trGrid(Dm / 64, Dm / 64);          // (16, 16)
  const int castBlocks = (int)(BSD / 4 / 256);  // 8192

  // fp32 -> bf16 activation casts; fp32 -> bf16 [N][K] weight transposes
  cast_bf16<<<castBlocks, blk, 0, stream>>>(q, qb, (int)(BSD / 4));
  cast_bf16<<<castBlocks, blk, 0, stream>>>(k, kb2, (int)(BSD / 4));
  cast_bf16<<<castBlocks, blk, 0, stream>>>(v, vb2, (int)(BSD / 4));
  transpose_cast_bf16<<<trGrid, blk, 0, stream>>>(Wq, Wqt);
  transpose_cast_bf16<<<trGrid, blk, 0, stream>>>(Wk, Wkt);
  transpose_cast_bf16<<<trGrid, blk, 0, stream>>>(Wv, Wvt);
  transpose_cast_bf16<<<trGrid, blk, 0, stream>>>(Wc, Wct);

  // q/k/v projections (bias + norm scale fused; v unscaled)
  gemm_bf16_wmma<<<gemmGrid, blk, 0, stream>>>(qb, Wqt, bq, NORM, qh);
  gemm_bf16_wmma<<<gemmGrid, blk, 0, stream>>>(kb2, Wkt, bk, NORM, kh);
  gemm_bf16_wmma<<<gemmGrid, blk, 0, stream>>>(vb2, Wvt, bv, 1.0f, vh);

  // depthwise causal conv + causal avg-pool, head split
  int cpBlocks = (int)((BHP + 255) / 256);  // 4096
  convpool<<<cpBlocks, blk, 0, stream>>>(qh, wcq, bcq, qp);
  convpool<<<cpBlocks, blk, 0, stream>>>(kh, wck, bck, kpp);
  convpool<<<cpBlocks, blk, 0, stream>>>(vh, wcv, bcv, vpp);

  // pooled causal attention + Wup projection
  attn_pool<<<Bsz * Hn, blk, 0, stream>>>(qp, kpp, vpp, Wup, bup, upd);

  // upsample + merge heads -> bf16 A for final projection
  upsample_merge_bf16<<<castBlocks, blk, 0, stream>>>(upd, mrg);

  // final projection
  gemm_bf16_wmma<<<gemmGrid, blk, 0, stream>>>(mrg, Wct, bc, 1.0f, (float*)d_out);
}
